// CustomMHA_1451698946646
// MI455X (gfx1250) — matrix operand order repaired and checked by reading, not compiled, not run on hardware
//
#include <hip/hip_runtime.h>

// ---------------------------------------------------------------------------
// CDNA5 / gfx1250 MHA forward. wave32, WMMA bf16 (fp32 accumulate),
// async global->LDS staging (ASYNCcnt) for transpose-free tiles.
//   d_out layout (flat f32, reference return order):
//     [0)            out   (8,1024,1024)      8,388,608
//     [8388608)      attn  (128,1024,1024)  134,217,728
//     [142606336)    k     (8,1024,1024)      8,388,608
//     [150994944)    v     (8,1024,1024)      8,388,608
//   d_ws: qbuf (32MB) + ctx (32MB)
// ---------------------------------------------------------------------------

typedef __attribute__((ext_vector_type(16))) __bf16 v16bf;
typedef __attribute__((ext_vector_type(8)))  __bf16 v8bf;
typedef __attribute__((ext_vector_type(8)))  float  v8f;

union FragA { v16bf v; v8bf h[2]; };

static __device__ __forceinline__ v8f wmma_bf16(v16bf a, v16bf b, v8f c) {
  // (neg_a, A, neg_b, B, c_mod, C, reuse_a, reuse_b)
  return __builtin_amdgcn_wmma_f32_16x16x32_bf16(false, a, false, b,
                                                 (short)0, c, false, false);
}

// Flat pointers into LDS carry the wave-relative LDS byte offset in their
// low 32 bits (flat->LDS aperture mapping truncates addr[31:0]).
static __device__ __forceinline__ uint32_t lds_addr_of(const void* p) {
  return (uint32_t)(uintptr_t)p;
}

// GLOBAL_LOAD_ASYNC_TO_LDS_B128 (GV mode): per-lane 16B global -> LDS copy,
// tracked with ASYNCcnt.  vdst = LDS byte address, vaddr = 64-bit VA.
static __device__ __forceinline__ void async_ld_b128(uint32_t lds, const void* g) {
  asm volatile("global_load_async_to_lds_b128 %0, %1, off"
               :: "v"(lds), "v"((unsigned long long)(uintptr_t)g)
               : "memory");
}
static __device__ __forceinline__ void wait_async0() {
  asm volatile("s_wait_asynccnt 0x0" ::: "memory");
}

// Build a 16x32 bf16 A-fragment for one lane from two contiguous 8-float runs
// staged in LDS: a[0..7] <- r0[0..7] (K=kb..kb+7), a[8..15] <- r1[0..7]
// (K=kb+16..kb+23).  cvts co-execute with the XDL WMMA pipe.
static __device__ __forceinline__ v16bf afrag_f32(const float* r0, const float* r1) {
  float4 x0 = *(const float4*)(r0);
  float4 x1 = *(const float4*)(r0 + 4);
  float4 x2 = *(const float4*)(r1);
  float4 x3 = *(const float4*)(r1 + 4);
  v16bf a;
  a[0]  = (__bf16)x0.x; a[1]  = (__bf16)x0.y; a[2]  = (__bf16)x0.z; a[3]  = (__bf16)x0.w;
  a[4]  = (__bf16)x1.x; a[5]  = (__bf16)x1.y; a[6]  = (__bf16)x1.z; a[7]  = (__bf16)x1.w;
  a[8]  = (__bf16)x2.x; a[9]  = (__bf16)x2.y; a[10] = (__bf16)x2.z; a[11] = (__bf16)x2.w;
  a[12] = (__bf16)x3.x; a[13] = (__bf16)x3.y; a[14] = (__bf16)x3.z; a[15] = (__bf16)x3.w;
  return a;
}

// B-fragment (32x16): lane = k, 16 contiguous n values from LDS f32.
static __device__ __forceinline__ v16bf bfrag_f32(const float* r) {
  float4 x0 = *(const float4*)(r);
  float4 x1 = *(const float4*)(r + 4);
  float4 x2 = *(const float4*)(r + 8);
  float4 x3 = *(const float4*)(r + 12);
  v16bf b;
  b[0]  = (__bf16)x0.x; b[1]  = (__bf16)x0.y; b[2]  = (__bf16)x0.z; b[3]  = (__bf16)x0.w;
  b[4]  = (__bf16)x1.x; b[5]  = (__bf16)x1.y; b[6]  = (__bf16)x1.z; b[7]  = (__bf16)x1.w;
  b[8]  = (__bf16)x2.x; b[9]  = (__bf16)x2.y; b[10] = (__bf16)x2.z; b[11] = (__bf16)x2.w;
  b[12] = (__bf16)x3.x; b[13] = (__bf16)x3.y; b[14] = (__bf16)x3.z; b[15] = (__bf16)x3.w;
  return b;
}

// ---------------------------------------------------------------------------
// Generic C = A(Mx1024) * W(Nx1024)^T + bias.  128x128 block tile, BK=32.
// A tile: async fp32 -> LDS.  W tile: VGPR-transposed bf16 -> LDS.
// Output column n routed to d0/d1/d2 by n>>10 (QKV split); col = n & 1023.
// ---------------------------------------------------------------------------
__global__ __launch_bounds__(128)
void mha_gemm_wxT_bias(const float* __restrict__ A,
                       const float* __restrict__ W,
                       const float* __restrict__ bias,
                       float* __restrict__ d0,
                       float* __restrict__ d1,
                       float* __restrict__ d2)
{
  constexpr int K = 1024;
  __shared__ float  Asf[128][36];   // [m][k] fp32, stride 36 dw (conflict-free)
  __shared__ __bf16 Bs[32][144];    // [k][n] bf16, 288B rows (32B aligned)

  const int tid  = threadIdx.x;
  const int lane = tid & 31;
  const int wv   = tid >> 5;
  const int l15  = lane & 15;
  const int half = lane >> 4;
  const int waveM = (wv >> 1) * 64;
  const int waveN = (wv & 1) * 64;
  const size_t mBase = (size_t)blockIdx.y * 128;
  const size_t nBase = (size_t)blockIdx.x * 128;
  const uint32_t asBase = lds_addr_of(&Asf[0][0]);

  v8f acc[4][4];
#pragma unroll
  for (int nt = 0; nt < 4; ++nt) {
    const float bv = bias[nBase + waveN + nt * 16 + l15];
    v8f c;
#pragma unroll
    for (int r = 0; r < 8; ++r) c[r] = bv;     // bias depends only on n
#pragma unroll
    for (int mt = 0; mt < 4; ++mt) acc[mt][nt] = c;
  }

  for (int k0 = 0; k0 < K; k0 += 32) {
    // Async-stage A tile: 128x32 f32 = 1024 B128 transfers, 8 per lane.
#pragma unroll
    for (int i = 0; i < 8; ++i) {
      const int t = tid + i * 128;
      const int r = t >> 3;
      const int c = (t & 7) * 4;
      async_ld_b128(asBase + (uint32_t)(r * 36 + c) * 4,
                    A + (mBase + r) * K + k0 + c);
    }
    // Stage W tile transposed into [k][n]: one n-column per thread.
    const float4* w4 = (const float4*)(W + (nBase + tid) * K + k0);
#pragma unroll
    for (int i = 0; i < 8; ++i) {
      float4 f = w4[i];
      Bs[i * 4 + 0][tid] = (__bf16)f.x; Bs[i * 4 + 1][tid] = (__bf16)f.y;
      Bs[i * 4 + 2][tid] = (__bf16)f.z; Bs[i * 4 + 3][tid] = (__bf16)f.w;
    }
    wait_async0();
    __syncthreads();

    FragA af[4]; v16bf bf_[4];
#pragma unroll
    for (int mt = 0; mt < 4; ++mt) {          // A: lane=m, two 8-elem K runs
      const int row = waveM + mt * 16 + l15;
      af[mt].v = afrag_f32(&Asf[row][half * 8], &Asf[row][16 + half * 8]);
    }
#pragma unroll
    for (int nt = 0; nt < 4; ++nt) {          // B: lane=n, K in table order
      for (int s = 0; s < 16; ++s)
        bf_[nt][s] = Bs[(s & 7) + half * 8 + (s >> 3) * 16][waveN + nt * 16 + l15];
    }
#pragma unroll
    for (int mt = 0; mt < 4; ++mt)
#pragma unroll
      for (int nt = 0; nt < 4; ++nt)
        acc[mt][nt] = wmma_bf16(af[mt].v, bf_[nt], acc[mt][nt]);
    __syncthreads();
  }

#pragma unroll
  for (int mt = 0; mt < 4; ++mt) {
#pragma unroll
    for (int nt = 0; nt < 4; ++nt) {
      const size_t ng = nBase + waveN + nt * 16 + l15;
      float* dst = (ng < 1024) ? d0 : (ng < 2048 ? d1 : d2);
      const size_t col  = ng & 1023;
      const size_t mrow = mBase + waveM + mt * 16 + half * 8;
      v8f c = acc[mt][nt];
#pragma unroll
      for (int r = 0; r < 8; ++r)
        dst[(mrow + r) * 1024 + col] = c[r];
    }
  }
}

// ---------------------------------------------------------------------------
// Scores: S[n,i,j] = (1/32) * sum_d Q[i,n,d]*K[j,n,d] + mask[n>>4,i,j]
// Q tile: async fp32 -> LDS.  K tile: VGPR-transposed bf16 -> LDS.
// ---------------------------------------------------------------------------
__global__ __launch_bounds__(128)
void mha_scores(const float* __restrict__ qbuf, const float* __restrict__ kbuf,
                const float* __restrict__ mask, float* __restrict__ attn,
                float scale)
{
  __shared__ float  Qsf[128][68];  // [i][d] fp32 (272B rows, 16B aligned)
  __shared__ __bf16 Ks[64][144];   // [d][j] bf16 (288B rows)

  const int tid = threadIdx.x, lane = tid & 31, wv = tid >> 5;
  const int l15 = lane & 15, half = lane >> 4;
  const int waveM = (wv >> 1) * 64, waveN = (wv & 1) * 64;
  const int n = blockIdx.z;
  const size_t iBase = (size_t)blockIdx.y * 128;
  const size_t jBase = (size_t)blockIdx.x * 128;
  const uint32_t qsBase = lds_addr_of(&Qsf[0][0]);

  // Async-stage Q tile: 128x64 f32 = 2048 B128 transfers, 16 per lane.
#pragma unroll
  for (int i = 0; i < 16; ++i) {
    const int t = tid + i * 128;
    const int r = t >> 4;
    const int c = (t & 15) * 4;
    async_ld_b128(qsBase + (uint32_t)(r * 68 + c) * 4,
                  qbuf + (iBase + r) * 8192 + n * 64 + c);
  }
  // Stage K tile transposed into [d][j].
  const float4* k4 = (const float4*)(kbuf + (jBase + tid) * 8192 + n * 64);
#pragma unroll
  for (int i = 0; i < 16; ++i) {
    float4 f = k4[i];
    Ks[i * 4 + 0][tid] = (__bf16)f.x; Ks[i * 4 + 1][tid] = (__bf16)f.y;
    Ks[i * 4 + 2][tid] = (__bf16)f.z; Ks[i * 4 + 3][tid] = (__bf16)f.w;
  }
  wait_async0();
  __syncthreads();

  v8f acc[4][4];
#pragma unroll
  for (int mt = 0; mt < 4; ++mt)
#pragma unroll
    for (int nt = 0; nt < 4; ++nt)
#pragma unroll
      for (int r = 0; r < 8; ++r) acc[mt][nt][r] = 0.f;

#pragma unroll
  for (int kk = 0; kk < 64; kk += 32) {
    FragA af[4]; v16bf bf_[4];
#pragma unroll
    for (int mt = 0; mt < 4; ++mt) {
      const int row = waveM + mt * 16 + l15;
      af[mt].v = afrag_f32(&Qsf[row][kk + half * 8], &Qsf[row][kk + 16 + half * 8]);
    }
#pragma unroll
    for (int nt = 0; nt < 4; ++nt)
      for (int s = 0; s < 16; ++s)
        bf_[nt][s] = Ks[kk + (s & 7) + half * 8 + (s >> 3) * 16][waveN + nt * 16 + l15];
#pragma unroll
    for (int mt = 0; mt < 4; ++mt)
#pragma unroll
      for (int nt = 0; nt < 4; ++nt)
        acc[mt][nt] = wmma_bf16(af[mt].v, bf_[nt], acc[mt][nt]);
  }

  const float* m0 = mask + ((size_t)(n >> 4) << 20);
  float*       a0 = attn + ((size_t)n << 20);
#pragma unroll
  for (int mt = 0; mt < 4; ++mt) {
#pragma unroll
    for (int nt = 0; nt < 4; ++nt) {
      const size_t i0 = iBase + waveM + mt * 16 + half * 8;
      const size_t j  = jBase + waveN + nt * 16 + l15;
      v8f c = acc[mt][nt];
#pragma unroll
      for (int r = 0; r < 8; ++r) {
        const size_t idx = (i0 + r) * 1024 + j;
        a0[idx] = c[r] * scale + m0[idx];
      }
    }
  }
}

// ---------------------------------------------------------------------------
// Row softmax over attn: 131072 rows x 1024. One wave32 per row.
// ---------------------------------------------------------------------------
__global__ __launch_bounds__(256)
void mha_softmax(float* __restrict__ attn)
{
  const int lane = threadIdx.x & 31;
  const int wv   = threadIdx.x >> 5;
  const size_t row = (size_t)blockIdx.x * 8 + wv;
  float4* p4 = (float4*)(attn + row * 1024);

  float4 v[8];
  float mx = -3.402823466e38f;
#pragma unroll
  for (int i = 0; i < 8; ++i) {
    v[i] = p4[lane + i * 32];
    mx = fmaxf(mx, fmaxf(fmaxf(v[i].x, v[i].y), fmaxf(v[i].z, v[i].w)));
  }
#pragma unroll
  for (int off = 16; off > 0; off >>= 1) mx = fmaxf(mx, __shfl_xor(mx, off, 32));

  float sum = 0.f;
#pragma unroll
  for (int i = 0; i < 8; ++i) {
    v[i].x = __expf(v[i].x - mx); v[i].y = __expf(v[i].y - mx);
    v[i].z = __expf(v[i].z - mx); v[i].w = __expf(v[i].w - mx);
    sum += v[i].x + v[i].y + v[i].z + v[i].w;
  }
#pragma unroll
  for (int off = 16; off > 0; off >>= 1) sum += __shfl_xor(sum, off, 32);

  const float inv = 1.0f / sum;
#pragma unroll
  for (int i = 0; i < 8; ++i) {
    v[i].x *= inv; v[i].y *= inv; v[i].z *= inv; v[i].w *= inv;
    p4[lane + i * 32] = v[i];
  }
}

// ---------------------------------------------------------------------------
// O[i,n,d] = sum_j P[n,i,j] * V[j,n,d].  Per head: M=1024, N=64, K=1024.
// Both tiles are transpose-free -> fully async staged as fp32.
// ---------------------------------------------------------------------------
__global__ __launch_bounds__(128)
void mha_pv(const float* __restrict__ attn, const float* __restrict__ vbuf,
            float* __restrict__ ctx)
{
  __shared__ float Psf[128][36];   // [i][j-chunk] fp32
  __shared__ float Vsf[32][68];    // [j][d] fp32

  const int tid = threadIdx.x, lane = tid & 31, wv = tid >> 5;
  const int l15 = lane & 15, half = lane >> 4;
  const int n = blockIdx.y;
  const size_t iBase = (size_t)blockIdx.x * 128;
  const float* P0 = attn + ((size_t)n << 20) + iBase * 1024;
  const uint32_t psBase = lds_addr_of(&Psf[0][0]);
  const uint32_t vsBase = lds_addr_of(&Vsf[0][0]);

  v8f acc[2][4];
#pragma unroll
  for (int mt = 0; mt < 2; ++mt)
#pragma unroll
    for (int nt = 0; nt < 4; ++nt)
#pragma unroll
      for (int r = 0; r < 8; ++r) acc[mt][nt][r] = 0.f;

  for (int j0 = 0; j0 < 1024; j0 += 32) {
    // P tile: 128x32 f32, 8 B128 per lane.
#pragma unroll
    for (int i = 0; i < 8; ++i) {
      const int t = tid + i * 128;
      const int r = t >> 3;
      const int c = (t & 7) * 4;
      async_ld_b128(psBase + (uint32_t)(r * 36 + c) * 4,
                    P0 + (size_t)r * 1024 + j0 + c);
    }
    // V tile: 32x64 f32, 4 B128 per lane.
#pragma unroll
    for (int i = 0; i < 4; ++i) {
      const int t = tid + i * 128;
      const int r = t >> 4;
      const int c = (t & 15) * 4;
      async_ld_b128(vsBase + (uint32_t)(r * 68 + c) * 4,
                    vbuf + (size_t)(j0 + r) * 8192 + n * 64 + c);
    }
    wait_async0();
    __syncthreads();

    FragA af[2]; v16bf bf_[4];
#pragma unroll
    for (int mt = 0; mt < 2; ++mt) {
      const int row = wv * 32 + mt * 16 + l15;
      af[mt].v = afrag_f32(&Psf[row][half * 8], &Psf[row][16 + half * 8]);
    }
#pragma unroll
    for (int nt = 0; nt < 4; ++nt)
      for (int s = 0; s < 16; ++s)
        bf_[nt][s] = (__bf16)Vsf[(s & 7) + half * 8 + (s >> 3) * 16][nt * 16 + l15];
#pragma unroll
    for (int mt = 0; mt < 2; ++mt)
#pragma unroll
      for (int nt = 0; nt < 4; ++nt)
        acc[mt][nt] = wmma_bf16(af[mt].v, bf_[nt], acc[mt][nt]);
    __syncthreads();
  }

#pragma unroll
  for (int mt = 0; mt < 2; ++mt) {
#pragma unroll
    for (int nt = 0; nt < 4; ++nt) {
      const size_t i0 = iBase + wv * 32 + mt * 16 + half * 8;
      const int d = nt * 16 + l15;
      v8f c = acc[mt][nt];
#pragma unroll
      for (int r = 0; r < 8; ++r)
        ctx[(i0 + r) * 8192 + n * 64 + d] = c[r];
    }
  }
}

// ---------------------------------------------------------------------------
extern "C" void kernel_launch(void* const* d_in, const int* in_sizes, int n_in,
                              void* d_out, int out_size, void* d_ws, size_t ws_size,
                              hipStream_t stream)
{
  (void)in_sizes; (void)n_in; (void)out_size; (void)ws_size;
  const float* query = (const float*)d_in[0];
  // d_in[1]=key, d_in[2]=value unused: self-attn projects from query only.
  const float* mask  = (const float*)d_in[3];
  const float* qkv_w = (const float*)d_in[4];
  const float* qkv_b = (const float*)d_in[5];
  const float* out_w = (const float*)d_in[6];
  const float* out_b = (const float*)d_in[7];

  float* out  = (float*)d_out;
  float* attn = out  + (size_t)8192 * 1024;           // 8,388,608
  float* kout = attn + (size_t)128 * 1024 * 1024;     // +134,217,728
  float* vout = kout + (size_t)8192 * 1024;

  float* qbuf = (float*)d_ws;                         // 32 MB
  float* ctx  = qbuf + (size_t)8192 * 1024;           // 32 MB

  // 1) QKV projection (M=8192,N=3072,K=1024); k/v columns written straight to d_out.
  mha_gemm_wxT_bias<<<dim3(24, 64), 128, 0, stream>>>(query, qkv_w, qkv_b,
                                                      qbuf, kout, vout);
  // 2) Scores + mask, scale = 1/sqrt(1024).
  mha_scores<<<dim3(8, 8, 128), 128, 0, stream>>>(qbuf, kout, mask, attn, 0.03125f);
  // 3) Row softmax in-place on the attn output.
  mha_softmax<<<dim3(16384), 256, 0, stream>>>(attn);
  // 4) P @ V -> context (scrambled layout == contiguous (8192,1024)).
  mha_pv<<<dim3(8, 128), 128, 0, stream>>>(attn, vout, ctx);
  // 5) Output projection (N=1024 -> all columns route to d0).
  mha_gemm_wxT_bias<<<dim3(8, 64), 128, 0, stream>>>(ctx, out_w, out_b,
                                                     out, out, out);
}
